// VMRN_26173530702302
// MI455X (gfx1250) — compile-verified, matrix operand order checked by reading, not run on hardware
//
#include <hip/hip_runtime.h>
#include <hip/hip_bf16.h>
#include <stdint.h>

#define USE_ASYNC_LDS 1

// ---------------- CDNA5 types ----------------
typedef __attribute__((ext_vector_type(16))) __bf16 v16bf;
typedef __attribute__((ext_vector_type(8)))  float  v8f;

#define DEV static __device__ __forceinline__

namespace {
constexpr int kC    = 512;   // backbone channels
constexpr int kFH   = 38, kFW = 38;
constexpr int kPix  = 49;    // 7x7 pooled pixels (valid)
constexpr int kPixP = 64;    // padded pixel rows (4 full 16-wide N tiles)
constexpr int kPairs = 528;  // B * N_OBJ*(N_OBJ-1) = 4*132
constexpr int kR    = 3 * kPairs;   // 1584 roi instances (o1 | o2 | union)
constexpr int kFCN  = 2048;
constexpr int kPairDim = 9408;      // 3 * 64 * 49
constexpr int kSegDim  = 3136;      // 64 * 49
constexpr float kScale = 1.0f / 16.0f;
constexpr float kEps   = 1e-5f;
constexpr int kLdsStride = 48;      // shorts per LDS B row (96B: 16B-aligned chunks, bank spread)
}

DEV unsigned short f2bf(float x) {
  union { float f; unsigned u; } c; c.f = x;
  unsigned r = 0x7FFFu + ((c.u >> 16) & 1u);
  return (unsigned short)((c.u + r) >> 16);
}
DEV int imax(int a, int b) { return a > b ? a : b; }
DEV int imin(int a, int b) { return a < b ? a : b; }

union FragBF { v16bf v; unsigned u[8]; };

// K-offset of fragment dword v (bf16 pair k,k+1) for lane group g.
DEV int kOff(int v, int g) { return (v < 4) ? (v * 2 + g * 8) : (16 + (v - 4) * 2 + g * 8); }

DEV v16bf fragContig(const unsigned short* base, int g) {
  FragBF f;
#pragma unroll
  for (int v = 0; v < 8; ++v)
    f.u[v] = *reinterpret_cast<const unsigned*>(base + kOff(v, g));
  return f.v;
}
DEV v16bf fragZero() {
  FragBF f;
#pragma unroll
  for (int v = 0; v < 8; ++v) f.u[v] = 0u;
  return f.v;
}

// ---------------- weight packing ----------------
__global__ void pack_bf16(const float* __restrict__ src, unsigned short* __restrict__ dst, int n) {
  int i = blockIdx.x * blockDim.x + threadIdx.x;
  if (i < n) dst[i] = f2bf(src[i]);
}

// [O][I][3][3] fp32 -> [O][9*I] bf16 with K order tap-major (t*I + c)
__global__ void pack_conv3(const float* __restrict__ src, unsigned short* __restrict__ dst, int O, int I) {
  int i = blockIdx.x * blockDim.x + threadIdx.x;
  if (i >= O * I * 9) return;
  int m = i / (I * 9);
  int rem = i - m * I * 9;
  int t = rem / I;
  int c = rem - t * I;
  dst[i] = f2bf(src[((size_t)m * I + c) * 9 + t]);
}

// [K][N] fp32 -> [N][K] bf16
__global__ void pack_transpose(const float* __restrict__ src, unsigned short* __restrict__ dst, int K, int N) {
  int i = blockIdx.x * blockDim.x + threadIdx.x;
  if (i >= K * N) return;
  int k = i / N, n = i - k * N;
  dst[(size_t)n * K + k] = f2bf(src[i]);
}

// ---------------- ROI align -> bf16 activations [R][64][512] (rows 49..63 zeroed) ----------------
__global__ __launch_bounds__(128)
void roi_align_bf16(const float* __restrict__ feat, const float* __restrict__ rois,
                    const int* __restrict__ roi_img, const int* __restrict__ po1,
                    const int* __restrict__ po2, unsigned short* __restrict__ X0) {
  int blk = blockIdx.x;              // r * 64 + pix
  int r = blk >> 6, pix = blk & 63;
  unsigned short* dst = X0 + ((size_t)r * kPixP + pix) * kC;
  if (pix >= kPix) {                 // zero padding rows so GEMMs can read unpredicated
    for (int c = threadIdx.x; c < kC; c += blockDim.x) dst[c] = 0;
    return;
  }
  int p = r % kPairs, seg = r / kPairs;
  int i1 = po1[p], i2 = po2[p];
  const float* bA = rois + (size_t)i1 * 4;
  const float* bB = rois + (size_t)i2 * 4;
  float x1, y1, x2, y2;
  if (seg == 0)      { x1 = bA[0]; y1 = bA[1]; x2 = bA[2]; y2 = bA[3]; }
  else if (seg == 1) { x1 = bB[0]; y1 = bB[1]; x2 = bB[2]; y2 = bB[3]; }
  else {
    x1 = fminf(bA[0], bB[0]); y1 = fminf(bA[1], bB[1]);
    x2 = fmaxf(bA[2], bB[2]); y2 = fmaxf(bA[3], bB[3]);
  }
  x1 *= kScale; y1 *= kScale; x2 *= kScale; y2 *= kScale;
  int id = roi_img[i1];
  int py = pix / 7, px = pix - py * 7;
  float gx = (px + 0.5f) / 7.0f, gy = (py + 0.5f) / 7.0f;
  float X = x1 + gx * (x2 - x1), Y = y1 + gy * (y2 - y1);
  float xf = floorf(X), yf = floorf(Y);
  float fx = X - xf, fy = Y - yf;
  int x0 = imin(imax((int)xf, 0), kFW - 1);
  int x1i = imin(x0 + 1, kFW - 1);
  int y0 = imin(imax((int)yf, 0), kFH - 1);
  int y1i = imin(y0 + 1, kFH - 1);
  float w00 = (1 - fy) * (1 - fx), w01 = (1 - fy) * fx, w10 = fy * (1 - fx), w11 = fy * fx;
  size_t fb = (size_t)id * kC * kFH * kFW;
  for (int c = threadIdx.x; c < kC; c += blockDim.x) {
    const float* ch = feat + fb + (size_t)c * kFH * kFW;
    float v = ch[y0 * kFW + x0] * w00 + ch[y0 * kFW + x1i] * w01 +
              ch[y1i * kFW + x0] * w10 + ch[y1i * kFW + x1i] * w11;
    dst[c] = f2bf(v);
  }
}

// ---------------- 1x1 conv: wave computes 16m x 64n, no bounds checks ----------------
// X: [R][64][inC] bf16, Wp: [3][outC][inC] bf16 -> Y: [R][64][outC] bf16 (bias+ReLU)
__global__ __launch_bounds__(32)
void conv1x1_wmma(const unsigned short* __restrict__ X, const unsigned short* __restrict__ Wp,
                  const float* __restrict__ bseg0, const float* __restrict__ bseg1,
                  const float* __restrict__ bseg2, unsigned short* __restrict__ Y,
                  int inC, int outC) {
  int r  = blockIdx.x;
  int m0 = blockIdx.y * 16;
  int seg = r / kPairs;
  const unsigned short* Xr = X + (size_t)r * kPixP * inC;
  const unsigned short* W  = Wp + (size_t)seg * outC * inC;
  const float* bias = (seg == 0) ? bseg0 : (seg == 1) ? bseg1 : bseg2;
  int lane = threadIdx.x & 31, li = lane & 15, g = lane >> 4;
  const unsigned short* Arow = W + (size_t)(m0 + li) * inC;
  v8f acc[4] = {};
  for (int k0 = 0; k0 < inC; k0 += 32) {
    v16bf a = fragContig(Arow + k0, g);
    v16bf bf[4];
#pragma unroll
    for (int t = 0; t < 4; ++t)   // batch all B loads -> staggered loadcnt waits
      bf[t] = fragContig(Xr + (size_t)(t * 16 + li) * inC + k0, g);
#pragma unroll
    for (int t = 0; t < 4; ++t)
      acc[t] = __builtin_amdgcn_wmma_f32_16x16x32_bf16(false, a, false, bf[t], (short)0, acc[t], false, false);
  }
#pragma unroll
  for (int t = 0; t < 4; ++t) {
    int n = t * 16 + li;
#pragma unroll
    for (int rr = 0; rr < 8; ++rr) {
      int m = m0 + rr + g * 8;
      float v = acc[t][rr] + bias[m];
      v = v > 0.0f ? v : 0.0f;
      Y[(size_t)r * kPixP * outC + (size_t)n * outC + m] = f2bf(v);
    }
  }
}

// ---------------- 3x3 conv (implicit im2col, pad=1): wave computes 16m x 64n ----------------
__global__ __launch_bounds__(32)
void conv3x3_wmma(const unsigned short* __restrict__ X, const unsigned short* __restrict__ Wp,
                  const float* __restrict__ bseg0, const float* __restrict__ bseg1,
                  const float* __restrict__ bseg2, unsigned short* __restrict__ Y,
                  int inC, int outC, int chanMajorOut) {
  int r  = blockIdx.x;
  int m0 = blockIdx.y * 16;
  int seg = r / kPairs;
  int K = 9 * inC;
  const unsigned short* Xr = X + (size_t)r * kPixP * inC;
  const unsigned short* W  = Wp + (size_t)seg * outC * K;
  const float* bias = (seg == 0) ? bseg0 : (seg == 1) ? bseg1 : bseg2;
  int lane = threadIdx.x & 31, li = lane & 15, g = lane >> 4;
  int ny[4], nx[4];
#pragma unroll
  for (int t = 0; t < 4; ++t) {
    int n = t * 16 + li;
    ny[t] = n / 7; nx[t] = n - ny[t] * 7;
  }
  const unsigned short* Arow = W + (size_t)(m0 + li) * K;
  v8f acc[4] = {};
  for (int k0 = 0; k0 < K; k0 += 32) {
    int tap = k0 / inC;           // 32 | inC -> whole step shares one tap
    int cb  = k0 - tap * inC;
    int dy = tap / 3 - 1, dx = tap % 3 - 1;
    v16bf a = fragContig(Arow + k0, g);
    v16bf bf[4];
#pragma unroll
    for (int t = 0; t < 4; ++t) {
      int n = t * 16 + li;
      int py = ny[t] + dy, px = nx[t] + dx;
      bool ok = (n < kPix) && py >= 0 && py < 7 && px >= 0 && px < 7;
      bf[t] = ok ? fragContig(Xr + (size_t)(py * 7 + px) * inC + cb, g) : fragZero();
    }
#pragma unroll
    for (int t = 0; t < 4; ++t)
      acc[t] = __builtin_amdgcn_wmma_f32_16x16x32_bf16(false, a, false, bf[t], (short)0, acc[t], false, false);
  }
#pragma unroll
  for (int t = 0; t < 4; ++t) {
    int n = t * 16 + li;
#pragma unroll
    for (int rr = 0; rr < 8; ++rr) {
      int m = m0 + rr + g * 8;
      float v = acc[t][rr] + bias[m];
      v = v > 0.0f ? v : 0.0f;
      unsigned short h = f2bf(v);
      if (chanMajorOut) {
        if (n < kPix) Y[(size_t)r * (64 * kPix) + (size_t)m * kPix + n] = h; // [R][64][49]
      } else {
        Y[(size_t)r * kPixP * outC + (size_t)n * outC + m] = h;
      }
    }
  }
}

// ---------------- FC GEMM: 3 waves/WG, async-LDS-staged B tile (64n x 32k), 16x64 per wave ----------------
// A (segmented=1): Y4 as [3][528][3136] bf16; A (segmented=0): [528][K] bf16.
// B: WT [N][K] bf16. Out: H [528][N] fp32 (+ per-column bias).
__global__ __launch_bounds__(96)
void fc_wmma(const unsigned short* __restrict__ A, const unsigned short* __restrict__ WT,
             const float* __restrict__ bias, float* __restrict__ H, int K, int segmented) {
  __shared__ unsigned short Bs[64 * kLdsStride];
  int tid  = threadIdx.x;
  int wave = tid >> 5;
  int lane = tid & 31, li = lane & 15, g = lane >> 4;
  int n0 = blockIdx.x * 64;
  int mb = blockIdx.y * 48 + wave * 16;     // wave's 16-row M slice
  v8f acc[4] = {};
  for (int k0 = 0; k0 < K; k0 += 32) {
    __syncthreads();
    // cooperative stage of B tile: 64 rows x 32 bf16 (4KB) in 16B chunks
#if USE_ASYNC_LDS
    for (int idx = tid; idx < 256; idx += 96) {   // last sweep is wave-uniform (wave 2 drops out whole)
      int nl = idx >> 2, q = idx & 3;
      unsigned ldsOff = (unsigned)(size_t)(&Bs[nl * kLdsStride + q * 8]);
      const unsigned short* src = WT + (size_t)(n0 + nl) * K + k0 + q * 8;
      asm volatile("global_load_async_to_lds_b128 %0, %1, off"
                   :: "v"(ldsOff), "v"(src) : "memory");
    }
    asm volatile("s_wait_asynccnt 0x0" ::: "memory");
#else
    for (int idx = tid; idx < 256; idx += 96) {
      int nl = idx >> 2, q = idx & 3;
      *reinterpret_cast<uint4*>(&Bs[nl * kLdsStride + q * 8]) =
          *reinterpret_cast<const uint4*>(WT + (size_t)(n0 + nl) * K + k0 + q * 8);
    }
#endif
    __syncthreads();
    const unsigned short* Arow;
    if (segmented) {
      int seg = k0 / kSegDim;               // 3136 = 98*32 -> whole step in one segment
      int rem = k0 - seg * kSegDim;
      Arow = A + ((size_t)(seg * kPairs + mb + li)) * kSegDim + rem;
    } else {
      Arow = A + (size_t)(mb + li) * K + k0;
    }
    if (k0 + 32 < K) {                      // prefetch next K-slab of A
      __builtin_prefetch((const void*)(Arow + 32), 0, 1);
    }
    v16bf a = fragContig(Arow, g);
    v16bf bf[4];
#pragma unroll
    for (int t = 0; t < 4; ++t)
      bf[t] = fragContig(&Bs[(size_t)(t * 16 + li) * kLdsStride], g);
#pragma unroll
    for (int t = 0; t < 4; ++t)
      acc[t] = __builtin_amdgcn_wmma_f32_16x16x32_bf16(false, a, false, bf[t], (short)0, acc[t], false, false);
  }
#pragma unroll
  for (int t = 0; t < 4; ++t) {
    int n = n0 + t * 16 + li;
#pragma unroll
    for (int rr = 0; rr < 8; ++rr) {
      int m = mb + rr + g * 8;
      H[(size_t)m * kFCN + n] = acc[t][rr] + bias[n];
    }
  }
}

// ---------------- batch-norm (+ ReLU) over 528-row batch ----------------
__global__ void bn_relu(const float* __restrict__ X, const float* __restrict__ gamma,
                        const float* __restrict__ beta, unsigned short* __restrict__ outBF,
                        float* __restrict__ outF, int rows, int cols) {
  int c = blockIdx.x * blockDim.x + threadIdx.x;
  if (c >= cols) return;
  float s = 0.0f, s2 = 0.0f;
  for (int r = 0; r < rows; ++r) {
    float v = X[(size_t)r * cols + c];
    s += v; s2 += v * v;
  }
  float mu = s / rows;
  float var = s2 / rows - mu * mu;
  float gg = gamma[c] * rsqrtf(var + kEps);
  float bb = beta[c];
  for (int r = 0; r < rows; ++r) {
    float v = (X[(size_t)r * cols + c] - mu) * gg + bb;
    v = v > 0.0f ? v : 0.0f;
    if (outBF) outBF[(size_t)r * cols + c] = f2bf(v);
    if (outF)  outF[(size_t)r * cols + c] = v;
  }
}

// ---------------- classifier out + softmax/post ----------------
__global__ void out_layer(const float* __restrict__ H, const float* __restrict__ W,
                          const float* __restrict__ bias, float* __restrict__ score) {
  int idx = blockIdx.x * blockDim.x + threadIdx.x;
  if (idx >= kPairs * 3) return;
  int row = idx / 3, col = idx - row * 3;
  float s = bias[col];
  const float* h = H + (size_t)row * kFCN;
  for (int k = 0; k < kFCN; ++k) s += h[k] * W[k * 3 + col];
  score[idx] = s;
}

__global__ void softmax_post(const float* __restrict__ score, float* __restrict__ post) {
  int q = blockIdx.x * blockDim.x + threadIdx.x;
  if (q >= kPairs / 2) return;
  float p1[3], p2[3];
#pragma unroll
  for (int half = 0; half < 2; ++half) {
    const float* s = score + (size_t)(2 * q + half) * 3;
    float m = fmaxf(s[0], fmaxf(s[1], s[2]));
    float e0 = __expf(s[0] - m), e1 = __expf(s[1] - m), e2 = __expf(s[2] - m);
    float inv = 1.0f / (e0 + e1 + e2);
    float* p = half ? p2 : p1;
    p[0] = e0 * inv; p[1] = e1 * inv; p[2] = e2 * inv;
  }
  post[q * 3 + 0] = (p1[0] + p2[1]) * 0.5f;
  post[q * 3 + 1] = (p1[1] + p2[0]) * 0.5f;
  post[q * 3 + 2] = (p1[2] + p2[2]) * 0.5f;
}

// ---------------- launch ----------------
extern "C" void kernel_launch(void* const* d_in, const int* in_sizes, int n_in,
                              void* d_out, int out_size, void* d_ws, size_t ws_size,
                              hipStream_t stream) {
  const float* base_feat = (const float*)d_in[0];
  const float* rois      = (const float*)d_in[1];
  const int*   roi_img   = (const int*)d_in[2];
  const int*   po1       = (const int*)d_in[3];
  const int*   po2       = (const int*)d_in[4];
  const float *w1s[3], *b1s[3], *w2s[3], *b2s[3], *w3s[3], *b3s[3], *w4s[3], *b4s[3];
  for (int seg = 0; seg < 3; ++seg) {
    int base = 5 + seg * 8;
    w1s[seg] = (const float*)d_in[base + 0]; b1s[seg] = (const float*)d_in[base + 1];
    w2s[seg] = (const float*)d_in[base + 2]; b2s[seg] = (const float*)d_in[base + 3];
    w3s[seg] = (const float*)d_in[base + 4]; b3s[seg] = (const float*)d_in[base + 5];
    w4s[seg] = (const float*)d_in[base + 6]; b4s[seg] = (const float*)d_in[base + 7];
  }
  const float* fc1_w = (const float*)d_in[29];
  const float* fc1_b = (const float*)d_in[30];
  const float* bn1_g = (const float*)d_in[31];
  const float* bn1_b = (const float*)d_in[32];
  const float* fc2_w = (const float*)d_in[33];
  const float* fc2_b = (const float*)d_in[34];
  const float* bn2_g = (const float*)d_in[35];
  const float* bn2_b = (const float*)d_in[36];
  const float* out_w = (const float*)d_in[37];
  const float* out_b = (const float*)d_in[38];

  char* base = (char*)d_ws;
  size_t off = 0;
  auto alloc = [&](size_t bytes) -> void* {
    void* p = base + off;
    off += (bytes + 255) & ~(size_t)255;
    return p;
  };
  unsigned short* X0   = (unsigned short*)alloc((size_t)kR * kPixP * 512 * 2);
  unsigned short* Y1   = (unsigned short*)alloc((size_t)kR * kPixP * 128 * 2);
  unsigned short* Y2   = (unsigned short*)alloc((size_t)kR * kPixP * 256 * 2);
  unsigned short* Y3   = (unsigned short*)alloc((size_t)kR * kPixP * 64 * 2);
  unsigned short* Y4   = (unsigned short*)alloc((size_t)kR * kSegDim * 2);   // [3][528][3136]
  unsigned short* W1p  = (unsigned short*)alloc((size_t)3 * 128 * 512 * 2);
  unsigned short* W2p  = (unsigned short*)alloc((size_t)3 * 256 * 1152 * 2);
  unsigned short* W3p  = (unsigned short*)alloc((size_t)3 * 64 * 256 * 2);
  unsigned short* W4p  = (unsigned short*)alloc((size_t)3 * 64 * 576 * 2);
  unsigned short* F1WT = (unsigned short*)alloc((size_t)kFCN * kPairDim * 2);
  unsigned short* F2WT = (unsigned short*)alloc((size_t)kFCN * kFCN * 2);
  float*          H1   = (float*)alloc((size_t)kPairs * kFCN * 4);
  unsigned short* H1BF = (unsigned short*)alloc((size_t)kPairs * kFCN * 2);
  float*          H2   = (float*)alloc((size_t)kPairs * kFCN * 4);
  float*          H2N  = (float*)alloc((size_t)kPairs * kFCN * 4);
  (void)ws_size; (void)n_in; (void)in_sizes; (void)out_size;

  // ---- pack weights (bf16, GEMM-friendly K order) ----
  for (int seg = 0; seg < 3; ++seg) {
    pack_bf16 <<<(128 * 512 + 255) / 256, 256, 0, stream>>>(w1s[seg], W1p + (size_t)seg * 128 * 512, 128 * 512);
    pack_conv3<<<(256 * 128 * 9 + 255) / 256, 256, 0, stream>>>(w2s[seg], W2p + (size_t)seg * 256 * 1152, 256, 128);
    pack_bf16 <<<(64 * 256 + 255) / 256, 256, 0, stream>>>(w3s[seg], W3p + (size_t)seg * 64 * 256, 64 * 256);
    pack_conv3<<<(64 * 64 * 9 + 255) / 256, 256, 0, stream>>>(w4s[seg], W4p + (size_t)seg * 64 * 576, 64, 64);
  }
  pack_transpose<<<((size_t)kPairDim * kFCN + 255) / 256, 256, 0, stream>>>(fc1_w, F1WT, kPairDim, kFCN);
  pack_transpose<<<((size_t)kFCN * kFCN + 255) / 256, 256, 0, stream>>>(fc2_w, F2WT, kFCN, kFCN);

  // ---- ROI align -> padded bf16 activations ----
  roi_align_bf16<<<kR * kPixP, 128, 0, stream>>>(base_feat, rois, roi_img, po1, po2, X0);

  // ---- conv pipes as WMMA GEMMs (wave tile 16m x 64n) ----
  conv1x1_wmma<<<dim3(kR, 128 / 16), 32, 0, stream>>>(X0, W1p, b1s[0], b1s[1], b1s[2], Y1, 512, 128);
  conv3x3_wmma<<<dim3(kR, 256 / 16), 32, 0, stream>>>(Y1, W2p, b2s[0], b2s[1], b2s[2], Y2, 128, 256, 0);
  conv1x1_wmma<<<dim3(kR, 64 / 16), 32, 0, stream>>>(Y2, W3p, b3s[0], b3s[1], b3s[2], Y3, 256, 64);
  conv3x3_wmma<<<dim3(kR, 64 / 16), 32, 0, stream>>>(Y3, W4p, b4s[0], b4s[1], b4s[2], Y4, 64, 64, 1);

  // ---- classifier (async-LDS-tiled WMMA GEMMs) ----
  fc_wmma<<<dim3(kFCN / 64, kPairs / 48), 96, 0, stream>>>(Y4, F1WT, fc1_b, H1, kPairDim, 1);
  bn_relu<<<(kFCN + 255) / 256, 256, 0, stream>>>(H1, bn1_g, bn1_b, H1BF, (float*)nullptr, kPairs, kFCN);
  fc_wmma<<<dim3(kFCN / 64, kPairs / 48), 96, 0, stream>>>(H1BF, F2WT, fc2_b, H2, kFCN, 0);
  bn_relu<<<(kFCN + 255) / 256, 256, 0, stream>>>(H2, bn2_g, bn2_b, (unsigned short*)nullptr, H2N, kPairs, kFCN);

  float* score = (float*)d_out;              // [528][3]
  float* post  = score + (size_t)kPairs * 3; // [264][3]
  out_layer   <<<(kPairs * 3 + 255) / 256, 256, 0, stream>>>(H2N, out_w, out_b, score);
  softmax_post<<<(kPairs / 2 + 255) / 256, 256, 0, stream>>>(score, post);
}
